// AttentiveManifoldMixer_69922067579277
// MI455X (gfx1250) — matrix operand order repaired and checked by reading, not compiled, not run on hardware
//
#include <hip/hip_runtime.h>

typedef __attribute__((ext_vector_type(16))) __bf16 v16bf;
typedef __attribute__((ext_vector_type(8)))  float  v8f;

#define HW        4096      // 64x64 pixels per batch
#define KCHUNKS   128       // main K = 4096 in chunks of 32 (i = kc>>1 fixed per chunk)
#define KCHTOT    130       // + 2 chunks of conv_b bias GEMM (K=64)
#define MT        4         // 4 M-tiles -> 64 output channels
#define NT        2         // 2 N-tiles per wave -> 32 pixels
#define WAVES     4         // 128 threads / WG

static __device__ __forceinline__ unsigned short f2bf(float f) {
  __bf16 h = (__bf16)f;
  return __builtin_bit_cast(unsigned short, h);
}

// ---------------- Kernel 1: SE attention -> s[b][64] --------------------
__global__ __launch_bounds__(256) void se_kernel(const float* __restrict__ x,
                                                 const float* __restrict__ fc1,
                                                 const float* __restrict__ fc2,
                                                 float* __restrict__ s) {
  int b = blockIdx.x, tid = threadIdx.x;
  int c = tid >> 2, q = tid & 3;
  const float* xp = x + ((size_t)b * 64 + c) * HW + q * 1024;
  float sum = 0.f;
  #pragma unroll 4
  for (int t = 0; t < 1024; t += 4) {
    float4 v = *(const float4*)(xp + t);
    sum += v.x + v.y + v.z + v.w;
  }
  __shared__ float part[256];
  __shared__ float meanv[64];
  __shared__ float yv[16];
  part[tid] = sum;
  __syncthreads();
  if (q == 0)
    meanv[c] = (part[c*4] + part[c*4+1] + part[c*4+2] + part[c*4+3]) * (1.0f/4096.0f);
  __syncthreads();
  if (tid < 16) {
    float a = 0.f;
    #pragma unroll
    for (int j = 0; j < 64; ++j) a += fc1[tid*64 + j] * meanv[j];
    yv[tid] = a > 0.f ? a : 0.f;
  }
  __syncthreads();
  if (tid < 64) {
    float a = 0.f;
    #pragma unroll
    for (int m = 0; m < 16; ++m) a += fc2[tid*16 + m] * yv[m];
    s[b*64 + tid] = 1.0f / (1.0f + __expf(-a));
  }
}

// ------- Kernel 2: repack conv_w (+conv_b) to bf16 WMMA A-fragment order -------
// A[c, i*64+j] = conv_w[c*64+i, j]; bias chunks kc=128,129: A[c,k'] = conv_b[c*64+k']
// Layout: Asw[((kc*MT + mt)*32 + lane)*16 + e]; 16-bit A 16x32 lane map (ISA 7.12.2):
//   K(e,lane) = (e&7) + (e>=8 ? 16:0) + (lane>=16 ? 8:0); M = mt*16 + (lane&15)
__global__ __launch_bounds__(256) void pack_a(const float* __restrict__ conv_w,
                                              const float* __restrict__ conv_b,
                                              unsigned short* __restrict__ Asw, int n) {
  int o = blockIdx.x * 256 + threadIdx.x;
  if (o >= n) return;
  int e    = o & 15;
  int lane = (o >> 4) & 31;
  int mt   = (o >> 9) & 3;
  int kc   = o >> 11;
  int K    = (e & 7) + ((e >> 3) << 4) + ((lane >> 4) << 3);
  int ch   = mt * 16 + (lane & 15);
  float v;
  if (kc < KCHUNKS) {
    int k = kc * 32 + K;                  // k = i*64 + j
    v = conv_w[(ch * 64 + (k >> 6)) * 64 + (k & 63)];
  } else {
    int kE = (kc - KCHUNKS) * 32 + K;     // bias GEMM: K' = i (0..63)
    v = conv_b[ch * 64 + kE];
  }
  Asw[o] = f2bf(v);
}

// ---------------- Kernel 3: fused bilinear mix via bf16 WMMA --------------------
__global__ __launch_bounds__(128) void mixer_main(const float* __restrict__ x,
                                                  const float* __restrict__ s,
                                                  const unsigned short* __restrict__ Asw,
                                                  float* __restrict__ out) {
  // per-wave staging tiles (each wave reads only its own slice)
  __shared__ __align__(16) unsigned int   xbr[WAVES][64][32]; // bf16(x) replicated in both halves, [ch][px]
  __shared__ __align__(16) unsigned short xat[WAVES][32][64]; // bf16(x*s), [px][ch]
  __shared__ __align__(16) unsigned short xtt[WAVES][32][64]; // bf16(x),   [px][ch]  (bias chunks)

  const int wave = threadIdx.x >> 5;
  const int lane = threadIdx.x & 31;
  const int wg   = blockIdx.x;              // 256 WGs x 128 px
  const int b    = wg >> 5;                 // 32 WGs per batch
  const int hw0  = (wg & 31) * 128 + wave * 32;
  const float* xb_base = x + (size_t)b * 64 * HW + hw0;

  // ---- stage x tiles: 64 ch x 32 px per wave ----
  #pragma unroll 4
  for (int ch = 0; ch < 64; ++ch) {
    float xv = xb_base[(size_t)ch * HW + lane];
    float sv = s[b * 64 + ch];
    unsigned short xu = f2bf(xv);
    xbr[wave][ch][lane] = (unsigned)xu * 0x10001u;  // replicate into both 16-bit halves
    xat[wave][lane][ch] = f2bf(xv * sv);
    xtt[wave][lane][ch] = xu;
  }
  __syncthreads();

  const int n    = lane & 15;               // B/C column within tile
  const int hiKB = (lane >> 4) << 3;        // B lane-half K offset (+8)

  // ---- preload xa fragments: per N-tile, per j-block (2 of them), 8 packed dwords ----
  // B 32x16 lane map: K(e,lane) = (e&7) + (e>=8 ? 16:0) + (lane>=16 ? 8:0); dword ep holds pair (2ep, 2ep+1)
  unsigned int xaf[NT][2][8];
  #pragma unroll
  for (int nt = 0; nt < NT; ++nt) {
    int pl = nt * 16 + n;
    #pragma unroll
    for (int jb = 0; jb < 2; ++jb)
      #pragma unroll
      for (int ep = 0; ep < 8; ++ep) {
        int jp = jb * 32 + 2 * (ep & 3) + ((ep >> 2) << 4) + hiKB;
        xaf[nt][jb][ep] = *(const unsigned int*)&xat[wave][pl][jp];
      }
  }

  v8f acc[MT][NT];
  #pragma unroll
  for (int mt = 0; mt < MT; ++mt)
    #pragma unroll
    for (int nt = 0; nt < NT; ++nt) { v8f z = {}; acc[mt][nt] = z; }

  union BU { unsigned int u[8]; v16bf v; };

  // ---- main GEMM sweep: K = 4096 in 128 chunks of 32; i constant per chunk ----
  for (int kc = 0; kc < KCHUNKS; ++kc) {
    const int i = kc >> 1, jb = kc & 1;
    const unsigned short* ap = Asw + (((size_t)(kc * MT) * 32 + lane) << 4);
    __builtin_prefetch(ap + (MT * 32 * 16), 0, 1);   // global_prefetch next A chunk
    v16bf afrag[MT];
    #pragma unroll
    for (int mt = 0; mt < MT; ++mt)
      afrag[mt] = *(const v16bf*)(ap + ((mt * 32) << 4));
    #pragma unroll
    for (int nt = 0; nt < NT; ++nt) {
      unsigned int xi = xbr[wave][i][nt * 16 + n];   // bf16(x[i,p]) in both halves
      BU bu;
      #pragma unroll
      for (int ep = 0; ep < 8; ++ep) {               // B = x[i] * xa[j-block]
        unsigned int r;
        asm("v_pk_mul_bf16 %0, %1, %2" : "=v"(r) : "v"(xi), "v"(xaf[nt][jb][ep]));
        bu.u[ep] = r;
      }
      #pragma unroll
      for (int mt = 0; mt < MT; ++mt)
        acc[mt][nt] = __builtin_amdgcn_wmma_f32_16x16x32_bf16(
            false, afrag[mt], false, bu.v, (short)0, acc[mt][nt], false, false);
    }
  }

  // ---- bias GEMM: out += conv_b(64x64) @ x, as 2 extra K-chunks ----
  #pragma unroll
  for (int kc = KCHUNKS; kc < KCHTOT; ++kc) {
    const unsigned short* ap = Asw + (((size_t)(kc * MT) * 32 + lane) << 4);
    v16bf afrag[MT];
    #pragma unroll
    for (int mt = 0; mt < MT; ++mt)
      afrag[mt] = *(const v16bf*)(ap + ((mt * 32) << 4));
    #pragma unroll
    for (int nt = 0; nt < NT; ++nt) {
      int pl = nt * 16 + n;
      BU bu;
      #pragma unroll
      for (int ep = 0; ep < 8; ++ep) {
        int jp = (kc - KCHUNKS) * 32 + 2 * (ep & 3) + ((ep >> 2) << 4) + hiKB;
        bu.u[ep] = *(const unsigned int*)&xtt[wave][pl][jp];
      }
      #pragma unroll
      for (int mt = 0; mt < MT; ++mt)
        acc[mt][nt] = __builtin_amdgcn_wmma_f32_16x16x32_bf16(
            false, afrag[mt], false, bu.v, (short)0, acc[mt][nt], false, false);
    }
  }

  // ---- epilogue: C/D layout -> (ch, px); lanes>=16 carry M+8 ----
  float* outb = out + (size_t)b * 64 * HW + hw0;
  const int mhi = (lane >> 4) << 3;
  #pragma unroll
  for (int mt = 0; mt < MT; ++mt)
    #pragma unroll
    for (int nt = 0; nt < NT; ++nt)
      #pragma unroll
      for (int r = 0; r < 8; ++r) {
        int ch = mt * 16 + r + mhi;
        outb[(size_t)ch * HW + nt * 16 + n] = acc[mt][nt][r];
      }
}

extern "C" void kernel_launch(void* const* d_in, const int* in_sizes, int n_in,
                              void* d_out, int out_size, void* d_ws, size_t ws_size,
                              hipStream_t stream) {
  (void)in_sizes; (void)n_in; (void)out_size; (void)ws_size;
  const float* x      = (const float*)d_in[0];
  const float* fc1_w  = (const float*)d_in[1];
  const float* fc2_w  = (const float*)d_in[2];
  const float* conv_w = (const float*)d_in[3];
  const float* conv_b = (const float*)d_in[4];
  float* out = (float*)d_out;

  float*          s_ws = (float*)d_ws;                              // 2 KB
  unsigned short* A_ws = (unsigned short*)((char*)d_ws + 4096);     // 520 KB bf16 packed A

  se_kernel<<<8, 256, 0, stream>>>(x, fc1_w, fc2_w, s_ws);

  const int nA = KCHTOT * MT * 32 * 16;                             // 266240
  pack_a<<<(nA + 255) / 256, 256, 0, stream>>>(conv_w, conv_b, A_ws, nA);

  mixer_main<<<256, 128, 0, stream>>>(x, s_ws, A_ws, out);
}
